// C2AENet_79087527789094
// MI455X (gfx1250) — matrix operand order repaired and checked by reading, not compiled, not run on hardware
//
#include <hip/hip_runtime.h>

// ---------------------------------------------------------------------------
// CDNA5 WMMA types + fragment loaders (contiguous K order k = 16*(lane/16) + i, same for A and B)
// ---------------------------------------------------------------------------
typedef __attribute__((ext_vector_type(16))) _Float16 v16h;
typedef __attribute__((ext_vector_type(8)))  _Float16 v8h;
typedef __attribute__((ext_vector_type(8)))  float    v8f;

// A fragment 16x32 f16: lane<16 -> row=lane, K=0..15;
// lane>=16 -> row=lane-16, K=16..31. Contiguous -> ds_load_b128.
static __device__ inline v16h frag_a_lds(const _Float16* s, int ld) {
  const int lane = threadIdx.x & 31;
  const int row  = lane & 15;
  const int k8   = (lane >> 4) << 4;
  v16h r;
#pragma unroll
  for (int h = 0; h < 16; ++h) {
    int k = h + k8;
    r[h] = s[row * ld + k];
  }
  return r;
}

// B fragment 32x16 f16 read from a TRANSPOSED LDS image (s[col*ld + k]):
// lane<16 -> col=lane, K=0..15; lane>=16 -> K=16..31. Contiguous -> ds_load_b128.
static __device__ inline v16h frag_b_lds_t(const _Float16* s, int ld) {
  const int lane = threadIdx.x & 31;
  const int col  = lane & 15;
  const int k16  = (lane >> 4) << 4;
  v16h r;
#pragma unroll
  for (int h = 0; h < 16; ++h) r[h] = s[col * ld + k16 + h];
  return r;
}

// ---------------------------------------------------------------------------
// KNN: fused distance + per-row top-16 (never materializes the NxN matrix)
// ---------------------------------------------------------------------------
__global__ __launch_bounds__(128)
void knn_kernel(const float* __restrict__ pts, int* __restrict__ idx, int N) {
  __shared__ float sx[128], sy[128], sz[128], sq[128];
  const int b = blockIdx.y;
  const int n = blockIdx.x * 128 + threadIdx.x;
  const float* base = pts + (size_t)b * N * 3;
  const float x = base[n * 3 + 0], y = base[n * 3 + 1], z = base[n * 3 + 2];
  const float qn = x * x + y * y + z * z;
  float bd[16]; int bi[16];
#pragma unroll
  for (int i = 0; i < 16; ++i) { bd[i] = 3.0e38f; bi[i] = 0; }
  for (int m0 = 0; m0 < N; m0 += 128) {
    const int mi = m0 + threadIdx.x;
    float lx = base[mi * 3 + 0], ly = base[mi * 3 + 1], lz = base[mi * 3 + 2];
    sx[threadIdx.x] = lx; sy[threadIdx.x] = ly; sz[threadIdx.x] = lz;
    sq[threadIdx.x] = lx * lx + ly * ly + lz * lz;
    __syncthreads();
    for (int t = 0; t < 128; ++t) {
      float d = qn + sq[t] - 2.f * (x * sx[t] + y * sy[t] + z * sz[t]);
      if (d < bd[15]) {
        int cand = m0 + t;
        int p = 15;
        while (p > 0 && bd[p - 1] > d) { bd[p] = bd[p - 1]; bi[p] = bi[p - 1]; --p; }
        bd[p] = d; bi[p] = cand;
      }
    }
    __syncthreads();
  }
#pragma unroll
  for (int i = 0; i < 16; ++i) idx[((size_t)b * N + n) * 16 + i] = bi[i];
}

// ---------------------------------------------------------------------------
// One-time per layer: build zero-padded, TRANSPOSED f16 weight image
// Wt[cout][Kpad] from f32 W[2cin][cout] (Kpad is a power of two).
// ---------------------------------------------------------------------------
__global__ void prep_wt_kernel(const float* __restrict__ W, _Float16* __restrict__ Wt,
                               int k2, int cout, int Kpad, int kpShift, int total) {
  int e = blockIdx.x * blockDim.x + threadIdx.x;
  if (e >= total) return;
  int col = e >> kpShift, kk = e & (Kpad - 1);
  Wt[e] = (kk < k2) ? (_Float16)W[(size_t)kk * cout + col] : (_Float16)0.f;
}

// ---------------------------------------------------------------------------
// Fused edge conv. Block = 8 points x 16 neighbors = 128 rows.
// The prepped f16 weight image is staged into LDS with
// GLOBAL_LOAD_ASYNC_TO_LDS_B128 (ASYNCcnt + s_wait_asynccnt), the edge-feature
// tile (128 x Kpad) is built in LDS, then the WMMA loop runs guard-free with
// zero global traffic. Total LDS <= 192KB of the 320KB WGP LDS.
// ---------------------------------------------------------------------------
__global__ __launch_bounds__(256)
void edge_conv_kernel(const float* __restrict__ h, const int* __restrict__ idx,
                      const _Float16* __restrict__ Wt, const float* __restrict__ bias,
                      float* __restrict__ out, int N, int cin, int cout, int Kpad) {
  extern __shared__ __align__(16) _Float16 sm[];
  _Float16* smA = sm;                       // 128 x Kpad
  _Float16* smW = sm + 128 * Kpad;          // cout x Kpad (transposed, padded)
  const int b   = blockIdx.y;
  const int n0  = blockIdx.x * 8;
  const int tid = threadIdx.x, lane = tid & 31, w = tid >> 5;
  const int k2  = 2 * cin;

  // stage W image: contiguous async copy, 16B per lane per issue
  {
    const int totalh = cout * Kpad;         // halves; multiple of 2048
    const unsigned ldsW = (unsigned)(unsigned long long)
        (__attribute__((address_space(3))) _Float16*)smW;
    for (int e = tid * 8; e < totalh; e += 2048) {
      unsigned d = ldsW + (unsigned)e * 2u;
      const _Float16* srcp = Wt + e;
      asm volatile("global_load_async_to_lds_b128 %0, %1, off"
                   :: "v"(d), "v"(srcp) : "memory");
    }
  }

  {   // build edge-feature tile; waves 0-3 write hi-part, waves 4-7 the diff
    const int row = tid & 127;
    const int which = tid >> 7;
    const int p = row >> 4, k = row & 15;
    const int n = n0 + p;
    const int j = idx[((size_t)b * N + n) * 16 + k];
    const float* hi = h + ((size_t)b * N + n) * cin;
    const float* hj = h + ((size_t)b * N + j) * cin;
    if (which == 0) {
      for (int c = 0; c < cin; ++c) smA[row * Kpad + c] = (_Float16)hi[c];
      for (int c = k2; c < Kpad; ++c) smA[row * Kpad + c] = (_Float16)0.f;
    } else {
      for (int c = 0; c < cin; ++c)
        smA[row * Kpad + cin + c] = (_Float16)(hj[c] - hi[c]);
    }
  }
  asm volatile("s_wait_asynccnt 0x0" ::: "memory");
  __syncthreads();

  const int nct = cout >> 4;     // cout is always a multiple of 16
  for (int ct = 0; ct < nct; ++ct) {
    v8f acc = {};
    for (int kc = 0; kc < Kpad; kc += 32) {
      v16h a  = frag_a_lds(&smA[(w * 16) * Kpad + kc], Kpad);
      v16h bf = frag_b_lds_t(&smW[(ct * 16) * Kpad + kc], Kpad);
      acc = __builtin_amdgcn_wmma_f32_16x16x32_f16(false, a, false, bf,
                                                   (short)0, acc, false, false);
    }
    // max over 16 neighbor rows: 8 in registers + cross-half via shfl_xor(16)
    float m = acc[0];
#pragma unroll
    for (int r = 1; r < 8; ++r) m = fmaxf(m, acc[r]);
    m = fmaxf(m, __shfl_xor(m, 16, 32));
    const int col = ct * 16 + (lane & 15);
    if (lane < 16)
      out[((size_t)b * N + n0 + w) * cout + col] = fmaxf(m + bias[col], 0.f);
  }
}

// ---------------------------------------------------------------------------
// Generic fused FC GEMM: Y = act(A@W + bias) (+ residual), f16 WMMA, f32 acc.
// REQUIRES M % 64 == 0 and Kd % 32 == 0 (true for every call site) -> the
// A path is guard-free and fully vectorized (float4 global, b128 LDS).
// B tile stored transposed so B fragments are contiguous ds_load_b128.
// ---------------------------------------------------------------------------
template <int ACT>   // 0 = none, 1 = relu
__global__ __launch_bounds__(256)
void fc_kernel(const float* __restrict__ A, const float* __restrict__ W,
               const float* __restrict__ bias, const float* __restrict__ residual,
               float* __restrict__ Y, int M, int Kd, int Cout) {
  constexpr int LDA = 40;     // halves; 80B rows -> 16B-aligned 8-half groups
  constexpr int LDBT = 40;
  __shared__ __align__(16) _Float16 As[64 * LDA];
  __shared__ __align__(16) _Float16 BsT[64 * LDBT];   // [col][k]
  const int m0 = blockIdx.x * 64;
  const int n0 = blockIdx.y * 64;
  const int tid = threadIdx.x, lane = tid & 31, w = tid >> 5;
  const int rt = w & 3, ch = w >> 2;
  v8f acc0 = {}, acc1 = {};
  for (int k0 = 0; k0 < Kd; k0 += 32) {
    {   // A: 64x32, one row-octet per thread, vectorized
      const int r = tid >> 2, c0 = (tid & 3) * 8;
      const float4* p =
          (const float4*)(A + (size_t)(m0 + r) * Kd + k0 + c0);
      float4 f0 = p[0], f1 = p[1];
      v8h hv;
      hv[0] = (_Float16)f0.x; hv[1] = (_Float16)f0.y;
      hv[2] = (_Float16)f0.z; hv[3] = (_Float16)f0.w;
      hv[4] = (_Float16)f1.x; hv[5] = (_Float16)f1.y;
      hv[6] = (_Float16)f1.z; hv[7] = (_Float16)f1.w;
      *(v8h*)(&As[r * LDA + c0]) = hv;
    }
#pragma unroll
    for (int i = 0; i < 8; ++i) {   // B: 32x64 -> transposed LDS image
      int e = tid + i * 256;
      int r = e >> 6, c = e & 63;   // r = k-offset, c = col-offset
      int gn = n0 + c;
      BsT[c * LDBT + r] = (gn < Cout) ? (_Float16)W[(size_t)(k0 + r) * Cout + gn]
                                      : (_Float16)0.f;
    }
    if (k0 + 32 < Kd)   // emits global_prefetch_b8
      __builtin_prefetch(A + (size_t)(m0 + (tid >> 2)) * Kd + k0 + 32, 0, 1);
    __syncthreads();
    v16h a  = frag_a_lds(&As[rt * 16 * LDA], LDA);
    v16h b0 = frag_b_lds_t(&BsT[(ch * 32) * LDBT], LDBT);
    v16h b1 = frag_b_lds_t(&BsT[(ch * 32 + 16) * LDBT], LDBT);
    acc0 = __builtin_amdgcn_wmma_f32_16x16x32_f16(false, a, false, b0, (short)0, acc0, false, false);
    acc1 = __builtin_amdgcn_wmma_f32_16x16x32_f16(false, a, false, b1, (short)0, acc1, false, false);
    __syncthreads();
  }
  const int colA  = n0 + ch * 32 + (lane & 15);
  const int colB  = colA + 16;
  const int rbase = m0 + rt * 16 + ((lane >> 4) << 3);
#pragma unroll
  for (int r = 0; r < 8; ++r) {
    int gm = rbase + r;
    if (colA < Cout) {
      float v = acc0[r] + bias[colA];
      if (ACT == 1) v = fmaxf(v, 0.f);
      if (residual) v += residual[(size_t)gm * Cout + colA];
      Y[(size_t)gm * Cout + colA] = v;
    }
    if (colB < Cout) {
      float v = acc1[r] + bias[colB];
      if (ACT == 1) v = fmaxf(v, 0.f);
      if (residual) v += residual[(size_t)gm * Cout + colB];
      Y[(size_t)gm * Cout + colB] = v;
    }
  }
}

// ---------------------------------------------------------------------------
// SE helpers + elementwise glue
// ---------------------------------------------------------------------------
__global__ void mean_kernel(const float* __restrict__ h, float* __restrict__ s,
                            int N, int c) {
  const int b = blockIdx.x, ch = threadIdx.x;
  if (ch >= c) return;
  float acc = 0.f;
  for (int n = 0; n < N; ++n) acc += h[((size_t)b * N + n) * c + ch];
  s[b * c + ch] = acc / (float)N;
}

__global__ void vec_fc_kernel(const float* __restrict__ s, const float* __restrict__ W,
                              const float* __restrict__ bias, float* __restrict__ out,
                              int Kd, int Cout, int act) {  // act: 1 relu, 2 sigmoid
  const int b = blockIdx.x, c = threadIdx.x;
  if (c >= Cout) return;
  float acc = bias[c];
  for (int k = 0; k < Kd; ++k) acc += s[b * Kd + k] * W[(size_t)k * Cout + c];
  if (act == 1) acc = fmaxf(acc, 0.f);
  else if (act == 2) acc = 1.f / (1.f + __expf(-acc));
  out[b * Cout + c] = acc;
}

__global__ void scale_kernel(float* __restrict__ h, const float* __restrict__ s,
                             int N, int c, int total) {
  int i = blockIdx.x * blockDim.x + threadIdx.x;
  if (i >= total) return;
  int b = i / (N * c), ch = i % c;
  h[i] *= s[b * c + ch];
}

__global__ void concat_kernel(const float* __restrict__ src, float* __restrict__ dst,
                              int rows, int c, int Ctot, int off, int total) {
  int i = blockIdx.x * blockDim.x + threadIdx.x;
  if (i >= total) return;
  int r = i / c, cc = i - r * c;
  dst[(size_t)r * Ctot + off + cc] = src[i];
}

__global__ void residual_kernel(float* __restrict__ x, const float* __restrict__ o,
                                float* __restrict__ dslice, int n) {
  int i = blockIdx.x * blockDim.x + threadIdx.x;
  if (i >= n) return;
  x[i] += o[i];
  dslice[i] = o[i];
}

// ---------------------------------------------------------------------------
// Host side: parameter index map mirroring setup_inputs() insertion order
// ---------------------------------------------------------------------------
struct PIdx { int W, b; };
struct Net {
  PIdx conv[4][3], sea[4][3], seb[4][3], cat[4], fuse[4], dl[4][4];
  PIdx e[3][6], D[3];
};
static Net make_net() {
  Net net; int k = 1;                 // input 0 is x
  auto nxt = [&](PIdx& p) { p.W = k++; p.b = k++; };
  for (int s = 0; s < 4; ++s) {
    for (int i = 0; i < 3; ++i) { nxt(net.conv[s][i]); nxt(net.sea[s][i]); nxt(net.seb[s][i]); }
    nxt(net.cat[s]);
    if (s > 0) nxt(net.fuse[s]);
    for (int j = 0; j < 4; ++j) nxt(net.dl[s][j]);
  }
  for (int j = 0; j < 3; ++j) { for (int i = 0; i < 6; ++i) nxt(net.e[j][i]); nxt(net.D[j]); }
  return net;
}

static inline void launch_fc(const float* A, const float* W, const float* b,
                             const float* res, float* Y, int M, int K, int C,
                             int act, hipStream_t st) {
  dim3 g((M + 63) / 64, (C + 63) / 64);
  if (act) fc_kernel<1><<<g, 256, 0, st>>>(A, W, b, res, Y, M, K, C);
  else     fc_kernel<0><<<g, 256, 0, st>>>(A, W, b, res, Y, M, K, C);
}

extern "C" void kernel_launch(void* const* d_in, const int* in_sizes, int n_in,
                              void* d_out, int out_size, void* d_ws, size_t ws_size,
                              hipStream_t stream) {
  (void)in_sizes; (void)n_in; (void)out_size; (void)ws_size;
  const int B = 2, N = 4096, BN = B * N;
  const int CIN[3]  = {3, 64, 128};
  const int OC[3]   = {64, 128, 256};
  const int KPAD[3] = {32, 128, 256};
  const int KPSH[3] = {5, 7, 8};      // log2(Kpad)
  Net net = make_net();
  auto F = [&](int i) { return (const float*)d_in[i]; };

  char* wp = (char*)d_ws;
  auto alloc = [&](size_t bytes) { void* p = (void*)wp; wp += (bytes + 255) & ~(size_t)255; return p; };

  float* xbuf = (float*)alloc((size_t)BN * 3 * 4);
  int*   idxb = (int*)alloc((size_t)BN * 16 * 4);
  float* outs[2][3]; float* dec[2][3]; float* fbuf[2];
  for (int s = 0; s < 2; ++s) {
    for (int i = 0; i < 3; ++i) outs[s][i] = (float*)alloc((size_t)BN * OC[i] * 4);
    for (int i = 0; i < 3; ++i) dec[s][i]  = (float*)alloc((size_t)BN * OC[2 - i] * 4);
    fbuf[s] = (float*)alloc((size_t)BN * 512 * 4);
  }
  float* t448  = (float*)alloc((size_t)BN * 448 * 4);
  float* t1024 = (float*)alloc((size_t)BN * 1024 * 4);
  float* tD    = (float*)alloc((size_t)BN * 512 * 4);
  float* tout  = (float*)alloc((size_t)BN * 3 * 4);
  float* sm = (float*)alloc((size_t)B * 256 * 4);
  float* s1 = (float*)alloc((size_t)B * 64 * 4);
  float* s2 = (float*)alloc((size_t)B * 256 * 4);
  _Float16* wtbuf[4][3];
  for (int s = 0; s < 4; ++s)
    for (int i = 0; i < 3; ++i)
      wtbuf[s][i] = (_Float16*)alloc((size_t)OC[i] * KPAD[i] * sizeof(_Float16));

  hipMemcpyAsync(xbuf, d_in[0], (size_t)BN * 3 * 4, hipMemcpyDeviceToDevice, stream);

  // one-time weight prep: transposed + zero-padded f16 images for all convs
  for (int s = 0; s < 4; ++s)
    for (int i = 0; i < 3; ++i) {
      int total = OC[i] * KPAD[i];
      prep_wt_kernel<<<(total + 255) / 256, 256, 0, stream>>>(
          F(net.conv[s][i].W), wtbuf[s][i], 2 * CIN[i], OC[i], KPAD[i], KPSH[i], total);
    }

  int cur = 0, prv = 1;
  for (int s = 0; s < 4; ++s) {
    knn_kernel<<<dim3(N / 128, B), 128, 0, stream>>>(xbuf, idxb, N);

    for (int i = 0; i < 3; ++i) {
      const float* hin = (i == 0) ? (const float*)xbuf : (const float*)outs[cur][i - 1];
      const int cin = CIN[i], cout = OC[i], Kpad = KPAD[i];
      const size_t shb = ((size_t)128 * Kpad + (size_t)cout * Kpad) * sizeof(_Float16);
      edge_conv_kernel<<<dim3(N / 8, B), 256, shb, stream>>>(
          hin, idxb, wtbuf[s][i], F(net.conv[s][i].b), outs[cur][i], N, cin, cout, Kpad);
      // SE block
      mean_kernel<<<B, 256, 0, stream>>>(outs[cur][i], sm, N, cout);
      vec_fc_kernel<<<B, 64, 0, stream>>>(sm, F(net.sea[s][i].W), F(net.sea[s][i].b), s1, cout, cout / 4, 1);
      vec_fc_kernel<<<B, 256, 0, stream>>>(s1, F(net.seb[s][i].W), F(net.seb[s][i].b), s2, cout / 4, cout, 2);
      {
        int tot = BN * cout;
        scale_kernel<<<(tot + 255) / 256, 256, 0, stream>>>(outs[cur][i], s2, N, cout, tot);
      }
      if (s > 0) {  // h = h + fc(enc_prev[i]) + fc(dec_prev[-i-1])
        launch_fc(outs[prv][i], F(net.e[s - 1][i].W), F(net.e[s - 1][i].b),
                  outs[cur][i], outs[cur][i], BN, cout, cout, 1, stream);
        launch_fc(dec[prv][2 - i], F(net.e[s - 1][5 - i].W), F(net.e[s - 1][5 - i].b),
                  outs[cur][i], outs[cur][i], BN, cout, cout, 1, stream);
      }
    }

    // decoder
    {
      int t0 = BN * 64, t1 = BN * 128, t2 = BN * 256;
      concat_kernel<<<(t0 + 255) / 256, 256, 0, stream>>>(outs[cur][0], t448, BN, 64, 448, 0, t0);
      concat_kernel<<<(t1 + 255) / 256, 256, 0, stream>>>(outs[cur][1], t448, BN, 128, 448, 64, t1);
      concat_kernel<<<(t2 + 255) / 256, 256, 0, stream>>>(outs[cur][2], t448, BN, 256, 448, 192, t2);
    }
    launch_fc(t448, F(net.cat[s].W), F(net.cat[s].b), nullptr, fbuf[cur], BN, 448, 512, 1, stream);
    if (s > 0) {
      launch_fc(fbuf[prv], F(net.D[s - 1].W), F(net.D[s - 1].b), nullptr, tD, BN, 512, 512, 1, stream);
      int tc = BN * 512;
      concat_kernel<<<(tc + 255) / 256, 256, 0, stream>>>(fbuf[cur], t1024, BN, 512, 1024, 0, tc);
      concat_kernel<<<(tc + 255) / 256, 256, 0, stream>>>(tD, t1024, BN, 512, 1024, 512, tc);
      launch_fc(t1024, F(net.fuse[s].W), F(net.fuse[s].b), nullptr, fbuf[cur], BN, 1024, 512, 1, stream);
    }
    launch_fc(fbuf[cur],    F(net.dl[s][0].W), F(net.dl[s][0].b), nullptr, dec[cur][0], BN, 512, 256, 1, stream);
    launch_fc(dec[cur][0],  F(net.dl[s][1].W), F(net.dl[s][1].b), nullptr, dec[cur][1], BN, 256, 128, 1, stream);
    launch_fc(dec[cur][1],  F(net.dl[s][2].W), F(net.dl[s][2].b), nullptr, dec[cur][2], BN, 128, 64, 1, stream);
    launch_fc(dec[cur][2],  F(net.dl[s][3].W), F(net.dl[s][3].b), nullptr, tout,        BN, 64, 3, 0, stream);

    residual_kernel<<<(BN * 3 + 255) / 256, 256, 0, stream>>>(
        xbuf, tout, (float*)d_out + (size_t)s * BN * 3, BN * 3);

    int t = cur; cur = prv; prv = t;
  }
}
